// GCN_429496729746
// MI455X (gfx1250) — compile-verified, matrix operand order checked by reading, not run on hardware
//
#include <hip/hip_runtime.h>
#include <hip/hip_bf16.h>

typedef __attribute__((ext_vector_type(16))) __bf16 v16bf;
typedef __attribute__((ext_vector_type(8)))  float  v8f;

#define D_FEAT 128
#define LDS_STRIDE 132   // 132 % 64 banks -> conflict-free column reads

// ---------------- degree / norm ----------------
__global__ void k_deg_init(float* __restrict__ deg, int n) {
    int i = blockIdx.x * blockDim.x + threadIdx.x;
    if (i < n) deg[i] = 1.0f;            // self-loop contributes 1
}

__global__ void k_deg_edges(const int* __restrict__ ei, int e, float* __restrict__ deg) {
    int i = blockIdx.x * blockDim.x + threadIdx.x;
    if (i < e) atomicAdd(&deg[ei[e + i]], 1.0f);   // dst = ei[E + i]
}

__global__ void k_dinv(const float* __restrict__ deg, float* __restrict__ dinv, int n) {
    int i = blockIdx.x * blockDim.x + threadIdx.x;
    if (i < n) dinv[i] = rsqrtf(deg[i]);           // deg >= 1 always
}

// ---------------- pack W into per-lane WMMA B-fragment layout (bf16 hi/lo) ----------------
// Fragment element j of lane L in (n_tile, k_tile) holds W[k][n] with
//   k = 32*kt + (j<8 ? j : j+8) + 8*(L>>4),  n = 16*nt + (L&15)
__global__ void k_wprep(const float* __restrict__ W, __bf16* __restrict__ hi, __bf16* __restrict__ lo) {
    int idx = blockIdx.x * blockDim.x + threadIdx.x;
    if (idx >= 8 * 4 * 32 * 16) return;
    int j    = idx & 15;
    int lane = (idx >> 4) & 31;
    int kt   = (idx >> 9) & 3;
    int nt   = idx >> 11;
    int k = kt * 32 + (j < 8 ? j : j + 8) + ((lane >> 4) * 8);
    int nn = nt * 16 + (lane & 15);
    float v = W[k * D_FEAT + nn];
    __bf16 h = (__bf16)v;
    hi[idx] = h;
    lo[idx] = (__bf16)(v - (float)h);
}

// ---------------- WMMA GEMM: H[n,128] = X[n,128] @ W[128,128], bf16x3 split ----------------
__global__ void __launch_bounds__(256) k_gemm(const float* __restrict__ X,
                                              const __bf16* __restrict__ Whi,
                                              const __bf16* __restrict__ Wlo,
                                              float* __restrict__ H, int n) {
    __shared__ float As[16 * LDS_STRIDE];
    const int tid = threadIdx.x;
    const int m0  = blockIdx.x * 16;

    // cooperative load of a 16x128 fp32 A-tile
    for (int i = tid; i < 16 * D_FEAT; i += 256) {
        int r = i >> 7, c = i & 127;
        int gr = m0 + r;
        As[r * LDS_STRIDE + c] = (gr < n) ? X[(size_t)gr * D_FEAT + c] : 0.0f;
    }
    __syncthreads();

    const int lane  = tid & 31;
    const int wv    = tid >> 5;          // n-tile 0..7
    const int ml    = lane & 15;         // A row within tile
    const int khalf = (lane >> 4) * 8;   // K half-select per WMMA layout

    v8f acc = {};
#pragma unroll
    for (int kt = 0; kt < 4; ++kt) {
        v16bf ahi, alo;
#pragma unroll
        for (int j = 0; j < 16; ++j) {
            int k = kt * 32 + (j < 8 ? j : j + 8) + khalf;
            float v = As[ml * LDS_STRIDE + k];
            __bf16 h = (__bf16)v;
            ahi[j] = h;
            alo[j] = (__bf16)(v - (float)h);
        }
        const v16bf bhi = *(const v16bf*)(Whi + (((size_t)wv * 4 + kt) * 32 + lane) * 16);
        const v16bf blo = *(const v16bf*)(Wlo + (((size_t)wv * 4 + kt) * 32 + lane) * 16);
        acc = __builtin_amdgcn_wmma_f32_16x16x32_bf16(false, ahi, false, bhi, (short)0, acc, false, false);
        acc = __builtin_amdgcn_wmma_f32_16x16x32_bf16(false, ahi, false, blo, (short)0, acc, false, false);
        acc = __builtin_amdgcn_wmma_f32_16x16x32_bf16(false, alo, false, bhi, (short)0, acc, false, false);
    }

    // C/D layout: VGPR r -> row r (+8 for upper half-wave), col = lane&15
    const int nbase = wv * 16 + (lane & 15);
    const int rofs  = (lane >> 4) * 8;
#pragma unroll
    for (int r = 0; r < 8; ++r) {
        int gr = m0 + r + rofs;
        if (gr < n) H[(size_t)gr * D_FEAT + nbase] = acc[r];
    }
}

// ---------------- zero ----------------
__global__ void k_zero(float* __restrict__ p, int n) {
    int i = blockIdx.x * blockDim.x + threadIdx.x;
    if (i < n) p[i] = 0.0f;
}

// ---------------- edge aggregation: one wave per edge, float4 per lane ----------------
__global__ void __launch_bounds__(256) k_agg(const int* __restrict__ ei, int e,
                                             const float* __restrict__ H,
                                             const float* __restrict__ dinv,
                                             float* __restrict__ agg) {
    long long t = (long long)blockIdx.x * blockDim.x + threadIdx.x;
    int edge = (int)(t >> 5);
    if (edge >= e) return;
    int lane = (int)t & 31;
    int s = ei[edge];
    int d = ei[e + edge];
    float w = dinv[s] * dinv[d];
    const float4 v = *(const float4*)(H + (size_t)s * D_FEAT + lane * 4);
    float* o = agg + (size_t)d * D_FEAT + lane * 4;
    atomicAdd(o + 0, v.x * w);
    atomicAdd(o + 1, v.y * w);
    atomicAdd(o + 2, v.z * w);
    atomicAdd(o + 3, v.w * w);
}

// ---------------- epilogue: self-loop + bias + ReLU ----------------
__global__ void k_post(const float* __restrict__ agg, const float* __restrict__ H,
                       const float* __restrict__ dinv, const float* __restrict__ b,
                       float* __restrict__ out, int n) {
    int t = blockIdx.x * blockDim.x + threadIdx.x;
    if (t >= n * D_FEAT) return;
    int node = t >> 7, f = t & 127;
    float di = dinv[node];
    float v = agg[t] + H[t] * di * di + b[f];
    out[t] = fmaxf(v, 0.0f);
}

extern "C" void kernel_launch(void* const* d_in, const int* in_sizes, int n_in,
                              void* d_out, int out_size, void* d_ws, size_t ws_size,
                              hipStream_t stream) {
    const float* x  = (const float*)d_in[0];
    const int*   ei = (const int*)d_in[1];
    const float* W1 = (const float*)d_in[2];
    const float* b1 = (const float*)d_in[3];
    const float* W2 = (const float*)d_in[4];
    const float* b2 = (const float*)d_in[5];
    float* out = (float*)d_out;

    const int n = in_sizes[0] / D_FEAT;   // 100000
    const int e = in_sizes[1] / 2;        // 1600000

    // carve workspace (256B-aligned slabs)
    char* ws = (char*)d_ws;
    size_t off = 0;
    auto carve = [&](size_t bytes) -> void* {
        void* p = ws + off;
        off = (off + bytes + 255) & ~(size_t)255;
        return p;
    };
    float*  hbuf   = (float*)carve((size_t)n * D_FEAT * sizeof(float));
    float*  aggbuf = (float*)carve((size_t)n * D_FEAT * sizeof(float));
    float*  deg    = (float*)carve((size_t)n * sizeof(float));
    float*  dinv   = (float*)carve((size_t)n * sizeof(float));
    const size_t wfrag = 8 * 4 * 32 * 16;               // 16384 elements
    __bf16* w1hi = (__bf16*)carve(wfrag * sizeof(__bf16));
    __bf16* w1lo = (__bf16*)carve(wfrag * sizeof(__bf16));
    __bf16* w2hi = (__bf16*)carve(wfrag * sizeof(__bf16));
    __bf16* w2lo = (__bf16*)carve(wfrag * sizeof(__bf16));

    const int nb_n   = (n + 255) / 256;
    const int nb_e   = (e + 255) / 256;
    const int nb_el  = (n * D_FEAT + 255) / 256;
    const int nb_m   = (n + 15) / 16;
    const int nb_agg = (int)(((long long)e * 32 + 255) / 256);

    // normalization
    k_deg_init <<<nb_n, 256, 0, stream>>>(deg, n);
    k_deg_edges<<<nb_e, 256, 0, stream>>>(ei, e, deg);
    k_dinv     <<<nb_n, 256, 0, stream>>>(deg, dinv, n);

    // weight packing (bf16 hi/lo fragments)
    k_wprep<<<64, 256, 0, stream>>>(W1, w1hi, w1lo);
    k_wprep<<<64, 256, 0, stream>>>(W2, w2hi, w2lo);

    // ----- layer 1 -----
    k_gemm<<<nb_m, 256, 0, stream>>>(x, w1hi, w1lo, hbuf, n);
    k_zero<<<nb_el, 256, 0, stream>>>(aggbuf, n * D_FEAT);
    k_agg <<<nb_agg, 256, 0, stream>>>(ei, e, hbuf, dinv, aggbuf);
    k_post<<<nb_el, 256, 0, stream>>>(aggbuf, hbuf, dinv, b1, aggbuf, n);  // in-place -> x1

    // ----- layer 2 -----
    k_gemm<<<nb_m, 256, 0, stream>>>(aggbuf, w2hi, w2lo, hbuf, n);
    k_zero<<<nb_el, 256, 0, stream>>>(aggbuf, n * D_FEAT);
    k_agg <<<nb_agg, 256, 0, stream>>>(ei, e, hbuf, dinv, aggbuf);
    k_post<<<nb_el, 256, 0, stream>>>(aggbuf, hbuf, dinv, b2, out, n);
}
